// FastRoutingLinear_9869834846649
// MI455X (gfx1250) — matrix operand order repaired and checked by reading, not compiled, not run on hardware
//
#include <hip/hip_runtime.h>
#include <hip/hip_bf16.h>

// ---------------------------------------------------------------------------
// FastRoutingLinear on MI455X (gfx1250, wave32):
//   logits GEMM in bf16 via v_wmma_f32_16x16x32_bf16 (fits weights in L2 as
//   bf16: 131 MB < 192 MB), fused per-row top-32 kept in wave registers
//   (one list entry per lane), exact f32 re-dot for the 32 winners.
// ---------------------------------------------------------------------------

typedef __attribute__((ext_vector_type(16))) __bf16 v16bf;
typedef __attribute__((ext_vector_type(8)))  float  v8f;

#define WAVE 32
#define K_DIM 2048
#define MTILE 16
#define NCHUNK 256
#define APAD 8
#define ASTRIDE (K_DIM + APAD)   // 2056 bf16 -> 4112B row stride, kills LDS bank conflicts

__device__ __forceinline__ float wave_reduce_add(float v) {
#pragma unroll
  for (int off = 16; off > 0; off >>= 1) v += __shfl_xor(v, off, WAVE);
  return v;
}
__device__ __forceinline__ float wave_reduce_min(float v) {
#pragma unroll
  for (int off = 16; off > 0; off >>= 1) v = fminf(v, __shfl_xor(v, off, WAVE));
  return v;
}

// ---------------------------------------------------------------------------
// Kernel 1: zero the dense output (512 MB) with wide stores.
// ---------------------------------------------------------------------------
__global__ void frl_zero_kernel(uint4* __restrict__ out, long long n4) {
  long long i = (long long)blockIdx.x * blockDim.x + threadIdx.x;
  if (i < n4) out[i] = make_uint4(0u, 0u, 0u, 0u);
}

// ---------------------------------------------------------------------------
// Kernel 2/3: per-row L2 normalize (f32 in) -> bf16 out.  One block per row.
// ---------------------------------------------------------------------------
__global__ __launch_bounds__(256)
void frl_rownorm_bf16_kernel(const float* __restrict__ src,
                             __bf16* __restrict__ dst, int K) {
  const int row = blockIdx.x;
  const float* p = src + (size_t)row * K;
  float ss = 0.f;
  for (int i = threadIdx.x; i < K; i += blockDim.x) {
    float v = p[i];
    ss = fmaf(v, v, ss);
  }
  ss = wave_reduce_add(ss);
  __shared__ float red[8];
  __shared__ float s_inv;
  const int wv = threadIdx.x >> 5, lane = threadIdx.x & 31;
  if (lane == 0) red[wv] = ss;
  __syncthreads();
  if (threadIdx.x == 0) {
    float t = 0.f;
    for (int i = 0; i < (int)(blockDim.x >> 5); ++i) t += red[i];
    s_inv = 1.0f / fmaxf(sqrtf(t), 1e-12f);
  }
  __syncthreads();
  const float inv = s_inv;
  __bf16* q = dst + (size_t)row * K;
  for (int i = threadIdx.x; i < K; i += blockDim.x)
    q[i] = (__bf16)(p[i] * inv);
}

// ---------------------------------------------------------------------------
// Kernel 4: cosine logits (bf16 WMMA) fused with per-row top-32 selection.
// Block = 16 rows x 256 threads (8 waves).  Wave w owns N-subtiles 2w, 2w+1
// of each 256-wide column chunk, and owns the top-32 lists of rows 2w, 2w+1
// (one (val,idx) entry per lane of the wave32).
// ---------------------------------------------------------------------------
__global__ __launch_bounds__(256)
void frl_cosine_topk_kernel(const __bf16* __restrict__ xn,
                            const __bf16* __restrict__ wn,
                            int* __restrict__ topk_out,
                            int OUT) {
  __shared__ __bf16 Atile[MTILE * ASTRIDE];        // 16 x 2048 (+pad) bf16
  __shared__ float  Llog[MTILE][NCHUNK];           // 16 x 256 f32 chunk logits

  const int tid  = threadIdx.x;
  const int wv   = tid >> 5;
  const int lane = tid & 31;
  const int m0   = blockIdx.x * MTILE;

  // ---- stage A tile into LDS (16 rows x 2048 bf16, uint4 chunks) ----
  for (int c = tid; c < MTILE * (K_DIM / 8); c += blockDim.x) {
    const int row = c >> 8;             // K_DIM/8 == 256 chunks per row
    const int k8  = (c & 255) << 3;
    *(uint4*)(&Atile[row * ASTRIDE + k8]) =
        *(const uint4*)(xn + (size_t)(m0 + row) * K_DIM + k8);
  }
  __syncthreads();

  // ---- top-32 state: one entry per lane, two rows per wave ----
  float tval[2] = {-INFINITY, -INFINITY};
  int   tidx[2] = {-1, -1};
  float thr[2]  = {-INFINITY, -INFINITY};

  // lane layouts used here (contiguous k, same for A and B; not the ISA table order):
  // A 16x32 bf16: lanes 0-15 -> M=lane, K kb+0..15 contiguous;
  //               lanes 16-31 -> M=lane-16, K kb+16..31.
  // B 32x16 bf16: lanes 0-15 -> N=lane, K kb+0..15 contiguous;
  //               lanes 16-31 -> N=lane-16, K kb+16..31.
  // C/D f32 16x16: lanes 0-15 -> N=lane, VGPR r -> M=r;
  //               lanes 16-31 -> N=lane-16, VGPR r -> M=r+8.
  const int mnA = lane & 15;
  const int khA = (lane >> 4) << 4;     // 0 or 16
  const int kbB = (lane >> 4) << 4;     // 0 or 16

  for (int n0 = 0; n0 < OUT; n0 += NCHUNK) {
    v8f acc0 = {};
    v8f acc1 = {};
    const int colA = n0 + (2 * wv) * 16 + mnA;
    const int colB = colA + 16;
    const __bf16* bpA = wn + (size_t)colA * K_DIM + kbB;
    const __bf16* bpB = wn + (size_t)colB * K_DIM + kbB;
    __builtin_prefetch(bpA, 0, 1);      // global_prefetch_b8
    __builtin_prefetch(bpB, 0, 1);

    for (int kb = 0; kb < K_DIM; kb += 32) {
      union { uint4 u[2]; v16bf v; } a, b0, b1;
      const __bf16* ap = &Atile[mnA * ASTRIDE + kb + khA];
      a.u[0]  = *(const uint4*)(ap);
      a.u[1]  = *(const uint4*)(ap + 8);
      b0.u[0] = *(const uint4*)(bpA + kb);
      b0.u[1] = *(const uint4*)(bpA + kb + 8);
      b1.u[0] = *(const uint4*)(bpB + kb);
      b1.u[1] = *(const uint4*)(bpB + kb + 8);
      acc0 = __builtin_amdgcn_wmma_f32_16x16x32_bf16(
          false, a.v, false, b0.v, (short)0, acc0, false, false);
      acc1 = __builtin_amdgcn_wmma_f32_16x16x32_bf16(
          false, a.v, false, b1.v, (short)0, acc1, false, false);
    }

    // ---- spill chunk logits to LDS ----
    {
      const int nn = lane & 15;
      const int rb = (lane >> 4) << 3;
#pragma unroll
      for (int r = 0; r < 8; ++r) {
        Llog[rb + r][(2 * wv) * 16 + nn]     = acc0[r];
        Llog[rb + r][(2 * wv + 1) * 16 + nn] = acc1[r];
      }
    }
    __syncthreads();

    // ---- top-32 update: ballot candidates above running min, argmin-replace
#pragma unroll
    for (int j = 0; j < 2; ++j) {
      const int r = 2 * wv + j;
      for (int it = 0; it < NCHUNK / 32; ++it) {
        const float v  = Llog[r][lane + (it << 5)];
        const int   ci = n0 + lane + (it << 5);
        unsigned long long mbits = __ballot((int)(v > thr[j]));
        while (mbits) {
          const int src = __builtin_ctzll(mbits);
          mbits &= mbits - 1;
          const float cv  = __shfl(v, src, WAVE);
          const int   cix = __shfl(ci, src, WAVE);
          if (cv > thr[j]) {
            float mv = tval[j];
            int   ml = lane;
#pragma unroll
            for (int off = 16; off > 0; off >>= 1) {
              const float ov = __shfl_xor(mv, off, WAVE);
              const int   ol = __shfl_xor(ml, off, WAVE);
              if (ov < mv || (ov == mv && ol < ml)) { mv = ov; ml = ol; }
            }
            if (cv > mv) {
              if (lane == ml) { tval[j] = cv; tidx[j] = cix; }
              thr[j] = wave_reduce_min(tval[j]);
            } else {
              thr[j] = mv;   // tighten threshold
            }
          }
        }
      }
    }
    __syncthreads();
  }

#pragma unroll
  for (int j = 0; j < 2; ++j) {
    const int r = 2 * wv + j;
    topk_out[(size_t)(m0 + r) * 32 + lane] = tidx[j];
  }
}

// ---------------------------------------------------------------------------
// Kernel 5: exact f32 dots for the 32 winners + bias, scatter into output.
// One block per row; 8 waves, each wave handles 4 of the 32 entries.
// ---------------------------------------------------------------------------
__global__ __launch_bounds__(256)
void frl_topk_scatter_kernel(const float* __restrict__ x,
                             const float* __restrict__ w,
                             const float* __restrict__ bias,
                             const int* __restrict__ topk,
                             float* __restrict__ out,
                             int K, int OUT) {
  const int row  = blockIdx.x;
  const int wv   = threadIdx.x >> 5;
  const int lane = threadIdx.x & 31;
  const float* xr = x + (size_t)row * K;
  for (int e = wv; e < 32; e += 8) {
    const int idx = topk[(size_t)row * 32 + e];
    const float* wr = w + (size_t)idx * K;
    float s = 0.f;
    for (int i = lane * 4; i < K; i += 128) {
      const float4 a = *(const float4*)(xr + i);
      const float4 b = *(const float4*)(wr + i);
      s = fmaf(a.x, b.x, s);
      s = fmaf(a.y, b.y, s);
      s = fmaf(a.z, b.z, s);
      s = fmaf(a.w, b.w, s);
    }
    s = wave_reduce_add(s);
    if (lane == 0) out[(size_t)row * OUT + idx] = s + bias[idx];
  }
}

// ---------------------------------------------------------------------------
extern "C" void kernel_launch(void* const* d_in, const int* in_sizes, int n_in,
                              void* d_out, int out_size, void* d_ws, size_t ws_size,
                              hipStream_t stream) {
  (void)n_in; (void)out_size; (void)ws_size;
  const float* x    = (const float*)d_in[0];   // [N, 2048] flattened
  const float* wgt  = (const float*)d_in[1];   // [32000, 2048]
  const float* bias = (const float*)d_in[2];   // [32000]
  float* out        = (float*)d_out;           // [N, 32000]

  const int K   = K_DIM;
  const int N   = in_sizes[0] / K;             // 4096
  const int OUT = in_sizes[2];                 // 32000

  // workspace layout: x_norm bf16 | w_norm bf16 | topk idx (~148 MB total)
  char* ws = (char*)d_ws;
  __bf16* xn = (__bf16*)ws;
  __bf16* wn = (__bf16*)(ws + (size_t)N * K * sizeof(__bf16));
  int* topk  = (int*)(ws + (size_t)(N + OUT) * K * sizeof(__bf16));

  // 1. zero dense output
  const long long n4 = ((long long)N * OUT) / 4;
  frl_zero_kernel<<<dim3((unsigned)((n4 + 255) / 256)), 256, 0, stream>>>(
      (uint4*)out, n4);

  // 2/3. row-normalize x and weight into bf16
  frl_rownorm_bf16_kernel<<<N, 256, 0, stream>>>(x, xn, K);
  frl_rownorm_bf16_kernel<<<OUT, 256, 0, stream>>>(wgt, wn, K);

  // 4. bf16 WMMA cosine logits + fused top-32
  frl_cosine_topk_kernel<<<N / MTILE, 256, 0, stream>>>(xn, wn, topk, OUT);

  // 5. exact f32 re-dot + bias, scatter
  frl_topk_scatter_kernel<<<N, 256, 0, stream>>>(x, wgt, bias, topk, out, K, OUT);
}
